// Layer_6923487281565
// MI455X (gfx1250) — matrix operand order repaired and checked by reading, not compiled, not run on hardware
//
#include <hip/hip_runtime.h>
#include <math.h>

// ---------------------------------------------------------------------------
// CDNA5 (gfx1250) transformer-encoder-layer forward.
// B=2, S=2048, D=1024, H=16, DH=64, FF=4096.
// Matmuls: v_wmma_f32_16x16x32_bf16 (bf16 in, f32 accumulate).
// GEMM tiles are DMA'd into LDS with TENSOR_LOAD_TO_LDS (TDM), double
// buffered and tracked with TENSORcnt.
// Workspace requirement: 152 MB (layout documented in kernel_launch).
// ---------------------------------------------------------------------------

typedef __bf16  bf16;
typedef __attribute__((ext_vector_type(16))) __bf16 v16bf;
typedef __attribute__((ext_vector_type(8)))  __bf16 v8bf;
typedef __attribute__((ext_vector_type(8)))  float  v8f;
typedef __attribute__((ext_vector_type(4)))  unsigned int v4u;
typedef __attribute__((ext_vector_type(8)))  int v8i;
typedef __attribute__((ext_vector_type(4)))  int v4i;

static __device__ __forceinline__ v8f wmma_bf16(v16bf a, v16bf b, v8f c) {
  // 8 args: (neg_a, A, neg_b, B, c_mod, C, reuse_a, reuse_b)
  return __builtin_amdgcn_wmma_f32_16x16x32_bf16(false, a, false, b,
                                                 (short)0, c, false, false);
}

// A fragment: 16x32 bf16 tile, LDS row-major [M][ldk].
// Layout used here (contiguous k, same as load_B_frag): lanes 0-15 / 16-31 both hold rows M=lane&15; elements 0..15 hold
// K = 16*half..+15 (contiguous bytes).
static __device__ __forceinline__ v16bf load_A_frag(const bf16* lds, int ldk,
                                                    int mbase, int kbase, int lane) {
  const bf16* row = lds + (size_t)(mbase + (lane & 15)) * ldk + kbase;
  int half = lane >> 4;
  union { v16bf v; v8bf h[2]; } u;
  u.h[0] = *(const v8bf*)(row + 16 * half);
  u.h[1] = *(const v8bf*)(row + 16 * half + 8);
  return u.v;
}

// B fragment: 32x16 bf16 tile staged N-major in LDS: ldsB[n][k] (ldk elems/row).
// Lane n = lane&15; lanes 0-15 hold K=0..15, lanes 16-31 hold K=16..31.
static __device__ __forceinline__ v16bf load_B_frag(const bf16* lds, int ldk,
                                                    int nbase, int kbase, int lane) {
  const bf16* row = lds + (size_t)(nbase + (lane & 15)) * ldk + kbase + 16 * (lane >> 4);
  union { v16bf v; v8bf h[2]; } u;
  u.h[0] = *(const v8bf*)(row);
  u.h[1] = *(const v8bf*)(row + 8);
  return u.v;
}

static __device__ __forceinline__ void cvt_store16(bf16* dst, const float* src, float scale) {
#pragma unroll
  for (int i = 0; i < 16; i += 4) {
    float4 f = *(const float4*)(src + i);
    dst[i + 0] = (bf16)(f.x * scale);
    dst[i + 1] = (bf16)(f.y * scale);
    dst[i + 2] = (bf16)(f.z * scale);
    dst[i + 3] = (bf16)(f.w * scale);
  }
}

// ---------------------------------------------------------------------------
// Tensor Data Mover: DMA a 2D bf16 tile [rows x cols] (row stride = `stride`
// elements) from global memory into LDS at byte offset `lds_addr`, packed
// row-major.  Descriptor packed per CDNA5 ISA ch.8 (D# groups 0/1; groups
// 2/3 zero for a 2D tensor).  Tracked with TENSORcnt.
// ---------------------------------------------------------------------------
static __device__ __forceinline__ void tdm_load_tile(
    unsigned lds_addr, const bf16* gsrc, unsigned rows, unsigned cols,
    unsigned long long stride_elems) {
  unsigned long long ga = (unsigned long long)(uintptr_t)gsrc;
  // group0: [1:0]=count=1, [63:32]=lds_addr, [120:64]=global_addr, [127:126]=type 2
  v4u g0 = { 1u,
             lds_addr,
             (unsigned)(ga & 0xFFFFFFFFull),
             (unsigned)((ga >> 32) & 0x01FFFFFFull) | 0x80000000u };
  unsigned td0 = (unsigned)stride_elems;  // tensor_dim0 >= tile cols -> never OOB
  unsigned td1 = rows;
  v8i g1 = {
    (int)0x10000u,                                               // data_size=1 (2 bytes)
    (int)((td0 & 0xFFFFu) << 16),                                // tensor_dim0[15:0]
    (int)(((td0 >> 16) & 0xFFFFu) | ((td1 & 0xFFFFu) << 16)),    // td0 hi | td1 lo
    (int)(((td1 >> 16) & 0xFFFFu) | ((cols & 0xFFFFu) << 16)),   // td1 hi | tile_dim0
    (int)(rows & 0xFFFFu),                                       // tile_dim1 | tile_dim2=0
    (int)(unsigned)(stride_elems & 0xFFFFFFFFull),               // tensor_dim0_stride lo
    (int)(unsigned)((stride_elems >> 32) & 0xFFFFull),           // stride hi | dim1_stride=0
    0
  };
  v4i gz = {0, 0, 0, 0};
#if defined(__clang_major__) && (__clang_major__ >= 23)
  v8i gz8 = {0, 0, 0, 0, 0, 0, 0, 0};
  __builtin_amdgcn_tensor_load_to_lds(g0, g1, gz, gz, gz8, 0);
#else
  __builtin_amdgcn_tensor_load_to_lds(g0, g1, gz, gz, 0);
#endif
}

// ---------------------------------------------------------------------------
// f32 -> bf16 bulk convert (n multiple of 4)
// ---------------------------------------------------------------------------
__global__ __launch_bounds__(256) void cvt_bf16_kernel(
    const float* __restrict__ in, bf16* __restrict__ out, int n) {
  int i = (blockIdx.x * 256 + threadIdx.x) * 4;
  if (i + 3 < n) {
    float4 f = *(const float4*)(in + i);
    out[i + 0] = (bf16)f.x;
    out[i + 1] = (bf16)f.y;
    out[i + 2] = (bf16)f.z;
    out[i + 3] = (bf16)f.w;
  }
}

// ---------------------------------------------------------------------------
// W[K][N] f32  ->  Wt[N][K] bf16   (32x32 LDS tiles; K,N multiples of 32)
// ---------------------------------------------------------------------------
__global__ __launch_bounds__(256) void transpose_cvt_kernel(
    const float* __restrict__ in, bf16* __restrict__ out, int K, int N) {
  __shared__ bf16 tile[32][33];
  const int k0 = blockIdx.x * 32, n0 = blockIdx.y * 32;
  const int tx = threadIdx.x & 31, ty = threadIdx.x >> 5;  // 8 rows per pass
#pragma unroll
  for (int i = 0; i < 32; i += 8)
    tile[ty + i][tx] = (bf16)in[(size_t)(k0 + ty + i) * N + n0 + tx];
  __syncthreads();
#pragma unroll
  for (int i = 0; i < 32; i += 8)
    out[(size_t)(n0 + ty + i) * K + k0 + tx] = tile[tx][ty + i];
}

// ---------------------------------------------------------------------------
// C[M,N] = act(A[M,K] @ Wt[N,K]^T + bias), A and Wt already bf16.
// Block tile 128x64, K-step 32.  128 threads = 4 waves; wave w owns rows
// [32w, 32w+32) of the tile (2 M-tiles x 4 N-tiles of WMMA accumulators).
// Tiles arrive via TDM into double-buffered LDS; wave 0 drives the DMA with
// s_wait_tensorcnt(2) so the next pair stays in flight during compute.
// ---------------------------------------------------------------------------
__global__ __launch_bounds__(128) void gemm_wmma_tdm_kernel(
    const bf16* __restrict__ A, const bf16* __restrict__ Bt,
    const float* __restrict__ bias, float* __restrict__ C,
    bf16* __restrict__ Cb, int M, int N, int K, int relu) {
  __shared__ bf16 lds_a[2][128][32];
  __shared__ bf16 lds_b[2][64][32];

  const int t = threadIdx.x, lane = t & 31, w = t >> 5;
  const int m0 = blockIdx.x * 128, n0 = blockIdx.y * 64;
  const int ksteps = K / 32;

  v8f acc[2][4];
#pragma unroll
  for (int i = 0; i < 2; ++i)
#pragma unroll
    for (int j = 0; j < 4; ++j) acc[i][j] = (v8f){0.f, 0.f, 0.f, 0.f, 0.f, 0.f, 0.f, 0.f};

  unsigned la[2] = { (unsigned)(uintptr_t)&lds_a[0][0][0],
                     (unsigned)(uintptr_t)&lds_a[1][0][0] };
  unsigned lb[2] = { (unsigned)(uintptr_t)&lds_b[0][0][0],
                     (unsigned)(uintptr_t)&lds_b[1][0][0] };

  // prologue: DMA tile pair for step 0 into buffer 0
  if (w == 0) {
    tdm_load_tile(la[0], A + (size_t)m0 * K, 128u, 32u, (unsigned long long)K);
    tdm_load_tile(lb[0], Bt + (size_t)n0 * K, 64u, 32u, (unsigned long long)K);
  }

  for (int s = 0; s < ksteps; ++s) {
    const int cur = s & 1, nxt = cur ^ 1;
    __syncthreads();  // everyone done reading buf[nxt] (previous step's data)
    if (w == 0) {
      if (s + 1 < ksteps) {
        const int k0n = (s + 1) * 32;
        tdm_load_tile(la[nxt], A + (size_t)m0 * K + k0n, 128u, 32u, (unsigned long long)K);
        tdm_load_tile(lb[nxt], Bt + (size_t)n0 * K + k0n, 64u, 32u, (unsigned long long)K);
        __builtin_amdgcn_s_wait_tensorcnt(2);  // oldest pair (buf[cur]) complete
      } else {
        __builtin_amdgcn_s_wait_tensorcnt(0);
      }
    }
    __syncthreads();  // buf[cur] visible to all waves

    v16bf bfrag[4];
#pragma unroll
    for (int nt = 0; nt < 4; ++nt)
      bfrag[nt] = load_B_frag(&lds_b[cur][0][0], 32, nt * 16, 0, lane);
#pragma unroll
    for (int i = 0; i < 2; ++i) {
      v16bf a = load_A_frag(&lds_a[cur][0][0], 32, w * 32 + i * 16, 0, lane);
#pragma unroll
      for (int nt = 0; nt < 4; ++nt)
        acc[i][nt] = wmma_bf16(a, bfrag[nt], acc[i][nt]);
    }
  }

  // Epilogue.  C element r of v8f -> row (r + 8*(lane>=16)), col lane&15.
  const int rr = 8 * (lane >> 4), cc = lane & 15;
#pragma unroll
  for (int i = 0; i < 2; ++i)
#pragma unroll
    for (int nt = 0; nt < 4; ++nt) {
      const int col = n0 + nt * 16 + cc;
      const float bv = bias ? bias[col] : 0.0f;
      const int rowb = m0 + w * 32 + i * 16 + rr;
#pragma unroll
      for (int r = 0; r < 8; ++r) {
        float v = acc[i][nt][r] + bv;
        if (relu) v = fmaxf(v, 0.0f);
        if (C)  C[(size_t)(rowb + r) * N + col] = v;
        if (Cb) Cb[(size_t)(rowb + r) * N + col] = (bf16)v;
      }
    }
}

// ---------------------------------------------------------------------------
// Flash attention: one block = (b, h, 64-query tile); 128 threads = 4 waves.
// Each wave owns 16 query rows; full DH=64 kept in 4 f32 accumulators.
// Writes f32 ctx and a bf16 shadow for the Wo GEMM.
// ---------------------------------------------------------------------------
__global__ __launch_bounds__(128) void flash_attn_kernel(
    const float* __restrict__ Q, const float* __restrict__ K,
    const float* __restrict__ V, const unsigned char* __restrict__ mask,
    float* __restrict__ O, bf16* __restrict__ Ob, int Bn, int S, int Hn) {
  __shared__ bf16  lds_q[64][64];   // [q][d]  (pre-scaled)
  __shared__ bf16  lds_k[64][64];   // [key][d]  == B^T layout for Q@K^T
  __shared__ bf16  lds_vt[64][64];  // [d][key]  == B layout for P@V
  __shared__ float lds_s[64][64];   // scores f32
  __shared__ bf16  lds_p[64][64];   // probabilities (A-layout for P@V)
  __shared__ float lds_m[64], lds_l[64], lds_al[64];

  const int t = threadIdx.x, lane = t & 31, w = t >> 5;
  const int ntiles = S / 64;
  const int qt = blockIdx.x % ntiles;
  const int h  = (blockIdx.x / ntiles) % Hn;
  const int b  = blockIdx.x / (ntiles * Hn);
  const int q0 = qt * 64;
  const int HD = Hn * 64;
  const float scale = 0.125f;  // 1/sqrt(64)

  {
    int row = t >> 1, ds = (t & 1) * 32;
    const float* src = Q + (size_t)(b * S + q0 + row) * HD + h * 64 + ds;
    cvt_store16(&lds_q[row][ds], src, scale);
    cvt_store16(&lds_q[row][ds + 16], src + 16, scale);
  }
  if (t < 64) { lds_m[t] = -3.0e38f; lds_l[t] = 0.0f; }

  v8f oacc[4];
#pragma unroll
  for (int nt = 0; nt < 4; ++nt) oacc[nt] = (v8f){0.f, 0.f, 0.f, 0.f, 0.f, 0.f, 0.f, 0.f};

  const int qb = w * 16;
  const int rr = 8 * (lane >> 4), cc = lane & 15;

  for (int j0 = 0; j0 < S; j0 += 64) {
    {
      int row = t >> 1, ds = (t & 1) * 32;
      const float* ks = K + (size_t)(b * S + j0 + row) * HD + h * 64 + ds;
      cvt_store16(&lds_k[row][ds], ks, 1.0f);
      cvt_store16(&lds_k[row][ds + 16], ks + 16, 1.0f);
      const float* vs = V + (size_t)(b * S + j0 + row) * HD + h * 64 + ds;
#pragma unroll
      for (int i = 0; i < 32; i += 4) {
        float4 f = *(const float4*)(vs + i);
        lds_vt[ds + i + 0][row] = (bf16)f.x;
        lds_vt[ds + i + 1][row] = (bf16)f.y;
        lds_vt[ds + i + 2][row] = (bf16)f.z;
        lds_vt[ds + i + 3][row] = (bf16)f.w;
      }
    }
    __syncthreads();

    // S = Q @ K^T  (two K=32 chunks over DH=64)
    v8f sacc[4];
#pragma unroll
    for (int nt = 0; nt < 4; ++nt) sacc[nt] = (v8f){0.f, 0.f, 0.f, 0.f, 0.f, 0.f, 0.f, 0.f};
#pragma unroll
    for (int dc = 0; dc < 64; dc += 32) {
      v16bf a = load_A_frag(&lds_q[0][0], 64, qb, dc, lane);
#pragma unroll
      for (int nt = 0; nt < 4; ++nt) {
        v16bf bb = load_B_frag(&lds_k[0][0], 64, nt * 16, dc, lane);
        sacc[nt] = wmma_bf16(a, bb, sacc[nt]);
      }
    }
#pragma unroll
    for (int nt = 0; nt < 4; ++nt)
#pragma unroll
      for (int r = 0; r < 8; ++r)
        lds_s[qb + rr + r][nt * 16 + cc] = sacc[nt][r];
    __syncthreads();

    // online softmax, one thread per query row
    if (t < 64) {
      const int row = t;
      const unsigned char* mrow = mask + (size_t)b * S + j0;
      float mold = lds_m[row], mnew = mold;
      for (int c = 0; c < 64; ++c) {
        float s = lds_s[row][c];
        if (mrow[c]) s = -3.0e38f;
        lds_s[row][c] = s;
        mnew = fmaxf(mnew, s);
      }
      float alpha = __expf(mold - mnew);
      float lsum = 0.0f;
      for (int c = 0; c < 64; ++c) {
        float s = lds_s[row][c];
        float p = (s < -1.0e37f) ? 0.0f : __expf(s - mnew);
        lds_p[row][c] = (bf16)p;
        lsum += p;
      }
      lds_l[row] = lds_l[row] * alpha + lsum;
      lds_m[row] = mnew;
      lds_al[row] = alpha;
    }
    __syncthreads();

    // rescale accumulators, then O += P @ V
    float al[8];
#pragma unroll
    for (int r = 0; r < 8; ++r) al[r] = lds_al[qb + rr + r];
#pragma unroll
    for (int nt = 0; nt < 4; ++nt)
#pragma unroll
      for (int r = 0; r < 8; ++r) oacc[nt][r] *= al[r];

#pragma unroll
    for (int kc = 0; kc < 64; kc += 32) {
      v16bf a = load_A_frag(&lds_p[0][0], 64, qb, kc, lane);
#pragma unroll
      for (int nt = 0; nt < 4; ++nt) {
        v16bf bb = load_B_frag(&lds_vt[0][0], 64, nt * 16, kc, lane);
        oacc[nt] = wmma_bf16(a, bb, oacc[nt]);
      }
    }
    __syncthreads();
  }

  float linv[8];
#pragma unroll
  for (int r = 0; r < 8; ++r) linv[r] = 1.0f / lds_l[qb + rr + r];
#pragma unroll
  for (int nt = 0; nt < 4; ++nt)
#pragma unroll
    for (int r = 0; r < 8; ++r) {
      const size_t idx = (size_t)(b * S + q0 + qb + rr + r) * HD + h * 64 + nt * 16 + cc;
      float v = oacc[nt][r] * linv[r];
      O[idx] = v;
      Ob[idx] = (bf16)v;
    }
}

// ---------------------------------------------------------------------------
// out = g * norm(a + b) + m, one 256-thread block per row; optional bf16 copy
// ---------------------------------------------------------------------------
__global__ __launch_bounds__(256) void add_ln_kernel(
    const float* __restrict__ a, const float* __restrict__ b,
    const float* __restrict__ g, const float* __restrict__ m,
    float* __restrict__ out, bf16* __restrict__ outb, int D) {
  __shared__ float xs[1024];
  __shared__ float red[256];
  const int row = blockIdx.x, t = threadIdx.x;
  const float* ar = a + (size_t)row * D;
  const float* br = b + (size_t)row * D;

  float s = 0.0f;
  for (int i = t; i < D; i += 256) {
    float v = ar[i] + br[i];
    xs[i] = v;
    s += v;
  }
  red[t] = s;
  __syncthreads();
  for (int off = 128; off > 0; off >>= 1) {
    if (t < off) red[t] += red[t + off];
    __syncthreads();
  }
  float mu = red[0] / (float)D;
  __syncthreads();

  float s2 = 0.0f;
  for (int i = t; i < D; i += 256) {
    float d = xs[i] - mu;
    s2 += d * d;
  }
  red[t] = s2;
  __syncthreads();
  for (int off = 128; off > 0; off >>= 1) {
    if (t < off) red[t] += red[t + off];
    __syncthreads();
  }
  float istd = rsqrtf(red[0] / (float)D + 1e-6f);

  for (int i = t; i < D; i += 256) {
    float v = g[i] * (xs[i] - mu) * istd + m[i];
    out[(size_t)row * D + i] = v;
    if (outb) outb[(size_t)row * D + i] = (bf16)v;
  }
}

// ---------------------------------------------------------------------------
extern "C" void kernel_launch(void* const* d_in, const int* in_sizes, int n_in,
                              void* d_out, int out_size, void* d_ws, size_t ws_size,
                              hipStream_t stream) {
  (void)in_sizes; (void)n_in; (void)out_size; (void)ws_size;
  const int Bn = 2, S = 2048, D = 1024, H = 16, FF = 4096;
  const int M = Bn * S;  // 4096

  const float*         x    = (const float*)d_in[0];
  const unsigned char* mask = (const unsigned char*)d_in[1];
  const float* Wq = (const float*)d_in[2];
  const float* Wk = (const float*)d_in[3];
  const float* Wv = (const float*)d_in[4];
  const float* Wo = (const float*)d_in[5];
  const float* bo = (const float*)d_in[6];
  const float* W1 = (const float*)d_in[7];
  const float* b1 = (const float*)d_in[8];
  const float* W2 = (const float*)d_in[9];
  const float* b2 = (const float*)d_in[10];
  const float* g1 = (const float*)d_in[11];
  const float* m1 = (const float*)d_in[12];
  const float* g2 = (const float*)d_in[13];
  const float* m2 = (const float*)d_in[14];
  float* out = (float*)d_out;

  // Workspace layout (byte offsets, 152 MB total).  Regions reused once their
  // producer/consumer lifetimes end:
  //   attn reuses q; x1 reuses k; x1b reuses v (first 8 MB).
  char* base = (char*)d_ws;
  const size_t MB = 1024 * 1024;
  bf16*  xb   = (bf16*)(base + 0 * MB);     //  8 MB  x in bf16
  bf16*  wqt  = (bf16*)(base + 8 * MB);     //  2 MB  Wq^T bf16 [N][K]
  bf16*  wkt  = (bf16*)(base + 10 * MB);    //  2 MB
  bf16*  wvt  = (bf16*)(base + 12 * MB);    //  2 MB
  bf16*  wot  = (bf16*)(base + 14 * MB);    //  2 MB
  bf16*  w1t  = (bf16*)(base + 16 * MB);    //  8 MB  W1^T [4096][1024]
  bf16*  w2t  = (bf16*)(base + 24 * MB);    //  8 MB  W2^T [1024][4096]
  float* q    = (float*)(base + 32 * MB);   // 16 MB
  float* k    = (float*)(base + 48 * MB);   // 16 MB
  float* v    = (float*)(base + 64 * MB);   // 16 MB
  float* ctx  = (float*)(base + 80 * MB);   // 16 MB
  bf16*  ctxb = (bf16*)(base + 96 * MB);    //  8 MB
  bf16*  ff1b = (bf16*)(base + 104 * MB);   // 32 MB  relu(x1@W1+b1) bf16
  float* ffo  = (float*)(base + 136 * MB);  // 16 MB
  float* attn = q;                          // q dead after attention
  float* x1   = k;                          // k dead after attention
  bf16*  x1b  = (bf16*)v;                   // v dead after attention

  // --- prep: bf16 conversions / weight transposes ---
  cvt_bf16_kernel<<<(M * D) / 1024, 256, 0, stream>>>(x, xb, M * D);
  transpose_cvt_kernel<<<dim3(D / 32, D / 32), 256, 0, stream>>>(Wq, wqt, D, D);
  transpose_cvt_kernel<<<dim3(D / 32, D / 32), 256, 0, stream>>>(Wk, wkt, D, D);
  transpose_cvt_kernel<<<dim3(D / 32, D / 32), 256, 0, stream>>>(Wv, wvt, D, D);
  transpose_cvt_kernel<<<dim3(D / 32, D / 32), 256, 0, stream>>>(Wo, wot, D, D);
  transpose_cvt_kernel<<<dim3(D / 32, FF / 32), 256, 0, stream>>>(W1, w1t, D, FF);
  transpose_cvt_kernel<<<dim3(FF / 32, D / 32), 256, 0, stream>>>(W2, w2t, FF, D);

  dim3 blk(128);
  // QKV projections
  gemm_wmma_tdm_kernel<<<dim3(M / 128, D / 64), blk, 0, stream>>>(
      xb, wqt, nullptr, q, nullptr, M, D, D, 0);
  gemm_wmma_tdm_kernel<<<dim3(M / 128, D / 64), blk, 0, stream>>>(
      xb, wkt, nullptr, k, nullptr, M, D, D, 0);
  gemm_wmma_tdm_kernel<<<dim3(M / 128, D / 64), blk, 0, stream>>>(
      xb, wvt, nullptr, v, nullptr, M, D, D, 0);
  // attention
  flash_attn_kernel<<<Bn * H * (S / 64), blk, 0, stream>>>(q, k, v, mask, ctx, ctxb, Bn, S, H);
  // output projection (writes into attn == old q region)
  gemm_wmma_tdm_kernel<<<dim3(M / 128, D / 64), blk, 0, stream>>>(
      ctxb, wot, bo, attn, nullptr, M, D, D, 0);
  // x1 = LN(attn + x)
  add_ln_kernel<<<M, 256, 0, stream>>>(attn, x, g1, m1, x1, x1b, D);
  // FFN
  gemm_wmma_tdm_kernel<<<dim3(M / 128, FF / 64), blk, 0, stream>>>(
      x1b, w1t, b1, nullptr, ff1b, M, FF, D, 1);
  gemm_wmma_tdm_kernel<<<dim3(M / 128, D / 64), blk, 0, stream>>>(
      ff1b, w2t, b2, ffo, nullptr, M, D, FF, 0);
  // out = LN(ffo + x1)
  add_ln_kernel<<<M, 256, 0, stream>>>(ffo, x1, g2, m2, out, nullptr, D);
}
